// EGCHUnit_27109833572723
// MI455X (gfx1250) — matrix operand order repaired and checked by reading, not compiled, not run on hardware
//
#include <hip/hip_runtime.h>
#include <hip/hip_bf16.h>
#include <math.h>

#define T_STEPS 8
#define NN 4096
#define DD 256

typedef __attribute__((ext_vector_type(16))) __bf16 v16bf;
typedef __attribute__((ext_vector_type(8)))  float  v8f;
typedef __attribute__((ext_vector_type(4)))  int    v4i;

union FragAB { uint4 q[2]; v16bf v; };

#if defined(__has_builtin)
#if __has_builtin(__builtin_amdgcn_global_load_async_to_lds_b128) && \
    __has_builtin(__builtin_amdgcn_s_wait_asynccnt)
#define USE_ASYNC 1
#endif
#endif
#ifndef USE_ASYNC
#define USE_ASYNC 0
#endif

#if USE_ASYNC
typedef __attribute__((address_space(1))) v4i as1_v4i;
typedef __attribute__((address_space(3))) v4i as3_v4i;
#endif

__device__ __forceinline__ unsigned short f32_to_bf16(float f) {
  unsigned int u = __float_as_uint(f);
  u += 0x7FFFu + ((u >> 16) & 1u);      // round-to-nearest-even
  return (unsigned short)(u >> 16);
}
__device__ __forceinline__ unsigned int pack_bf16x2(float a, float b) {
  return (unsigned int)f32_to_bf16(a) | ((unsigned int)f32_to_bf16(b) << 16);
}
__device__ __forceinline__ float sigmoidf_(float x) {
  return 1.0f / (1.0f + __expf(-x));
}
// 8 consecutive fp32 (LDS) -> 8 packed bf16 in one uint4
__device__ __forceinline__ void cvt8(const float* p, uint4& q) {
  float4 a = *(const float4*)p;
  float4 b = *(const float4*)(p + 4);
  q.x = pack_bf16x2(a.x, a.y); q.y = pack_bf16x2(a.z, a.w);
  q.z = pack_bf16x2(b.x, b.y); q.w = pack_bf16x2(b.z, b.w);
}

// ---------------------------------------------------------------------------
// Prep (once): p_norm_inv, W_ih/W_hh -> bf16, h0 = W0
// ---------------------------------------------------------------------------
__global__ void prep_kernel(const float* __restrict__ pool_p,
                            const float* __restrict__ W_ih,
                            const float* __restrict__ W_hh,
                            const float* __restrict__ W0,
                            float* __restrict__ pninv,
                            unsigned short* __restrict__ Wih_bf,
                            unsigned short* __restrict__ Whh_bf,
                            float* __restrict__ h) {
  int gid = blockIdx.x * 256 + threadIdx.x;
  if (blockIdx.x == 0) {
    __shared__ float red[256];
    float v = pool_p[threadIdx.x];
    red[threadIdx.x] = v * v;
    __syncthreads();
    for (int st = 128; st; st >>= 1) {
      if (threadIdx.x < st) red[threadIdx.x] += red[threadIdx.x + st];
      __syncthreads();
    }
    if (threadIdx.x == 0) pninv[0] = rsqrtf(red[0]);
  }
  for (int i = gid; i < 768 * DD; i += gridDim.x * 256) {
    Wih_bf[i] = f32_to_bf16(W_ih[i]);
    Whh_bf[i] = f32_to_bf16(W_hh[i]);
  }
  for (int i = gid; i < DD * DD; i += gridDim.x * 256) h[i] = W0[i];
}

// ---------------------------------------------------------------------------
// score[i] = dot(x[i], p) * pninv ; one wave32 per row
// ---------------------------------------------------------------------------
__global__ void score_kernel(const float* __restrict__ x,
                             const float* __restrict__ p,
                             const float* __restrict__ pninv,
                             float* __restrict__ s) {
  int w = (blockIdx.x * blockDim.x + threadIdx.x) >> 5;   // row 0..4095
  int lane = threadIdx.x & 31;
  const float* xr = x + (size_t)w * DD;
  float acc = 0.f;
  for (int k = lane; k < DD; k += 32) acc += xr[k] * p[k];
  for (int off = 16; off; off >>= 1) acc += __shfl_xor(acc, off, 32);
  if (lane == 0) s[w] = acc * pninv[0];
}

// ---------------------------------------------------------------------------
// exact top-k=256 via ranking (matches jax.lax.top_k: descending, stable ties)
// ---------------------------------------------------------------------------
__global__ void rank_kernel(const float* __restrict__ s,
                            int* __restrict__ sel, float* __restrict__ gate) {
  int i = blockIdx.x * 256 + threadIdx.x;
  float si = s[i];
  __shared__ float sh[256];
  int rank = 0;
  for (int base = 0; base < NN; base += 256) {
    sh[threadIdx.x] = s[base + threadIdx.x];
    __syncthreads();
    for (int j = 0; j < 256; ++j) {
      float sj = sh[j];
      rank += (sj > si) || (sj == si && (base + j) < i);
    }
    __syncthreads();
  }
  if (rank < DD) { sel[rank] = i; gate[rank] = tanhf(si); }
}

// ---------------------------------------------------------------------------
// GRU cell: hout = GRU(hin, x_hat). Block = 16 rows x 128 cols, ONE 16x16
// tile per wave (6 f32 accum tiles, no spill pressure). h double-buffered.
// Emits WT_bf16[col][row] = bf16(h_new), column-major B operand for y.
// ---------------------------------------------------------------------------
__global__ void __launch_bounds__(256)
gru_kernel(const float* __restrict__ x,
           const int* __restrict__ sel,
           const float* __restrict__ gate,
           const unsigned short* __restrict__ Wih,
           const unsigned short* __restrict__ Whh,
           const float* __restrict__ b_ih,
           const float* __restrict__ b_hh,
           const float* __restrict__ hin,
           float* __restrict__ hout,
           unsigned short* __restrict__ WT) {
  __shared__ unsigned short lx[16 * 264];
  __shared__ unsigned short lh[16 * 264];
  const int tid = threadIdx.x;
  const int rowband = blockIdx.x & 15;   // 16 bands of 16 rows
  const int colgrp  = blockIdx.x >> 4;   // 0..1 (128 cols each)
  {
    int row = tid >> 4;          // 0..15
    int kb  = (tid & 15) * 16;   // 0..240
    int grow = rowband * 16 + row;
    int srow = sel[grow];
    float g = gate[grow];
    const float* xr = x + (size_t)srow * DD + kb;
    const float* hr = hin + (size_t)grow * DD + kb;
    for (int i = 0; i < 16; i += 4) {
      float4 xv = *(const float4*)(xr + i);
      float4 hv = *(const float4*)(hr + i);
      *(uint2*)&lx[row * 264 + kb + i] =
          make_uint2(pack_bf16x2(xv.x * g, xv.y * g), pack_bf16x2(xv.z * g, xv.w * g));
      *(uint2*)&lh[row * 264 + kb + i] =
          make_uint2(pack_bf16x2(hv.x, hv.y), pack_bf16x2(hv.z, hv.w));
    }
  }
  __syncthreads();

  const int wave = tid >> 5, lane = tid & 31;
  const int mloc = lane & 15, half = lane >> 4;
  const int col = colgrp * 128 + wave * 16 + mloc;

  v8f ir = {}, iz = {}, in_ = {}, hr_ = {}, hz = {}, hn = {};
  for (int kc = 0; kc < DD; kc += 32) {
    FragAB ax, ah, b;
    const unsigned short* pax = &lx[mloc * 264 + kc + half * 16];
    ax.q[0] = *(const uint4*)pax; ax.q[1] = *(const uint4*)(pax + 8);
    const unsigned short* pah = &lh[mloc * 264 + kc + half * 16];
    ah.q[0] = *(const uint4*)pah; ah.q[1] = *(const uint4*)(pah + 8);
    const size_t kk = (size_t)kc + half * 16;
    { const uint4* p = (const uint4*)(Wih + (size_t)col * DD + kk);
      b.q[0] = p[0]; b.q[1] = p[1];
      ir = __builtin_amdgcn_wmma_f32_16x16x32_bf16(false, ax.v, false, b.v, (short)0, ir, false, false); }
    { const uint4* p = (const uint4*)(Wih + (size_t)(col + 256) * DD + kk);
      b.q[0] = p[0]; b.q[1] = p[1];
      iz = __builtin_amdgcn_wmma_f32_16x16x32_bf16(false, ax.v, false, b.v, (short)0, iz, false, false); }
    { const uint4* p = (const uint4*)(Wih + (size_t)(col + 512) * DD + kk);
      b.q[0] = p[0]; b.q[1] = p[1];
      in_ = __builtin_amdgcn_wmma_f32_16x16x32_bf16(false, ax.v, false, b.v, (short)0, in_, false, false); }
    { const uint4* p = (const uint4*)(Whh + (size_t)col * DD + kk);
      b.q[0] = p[0]; b.q[1] = p[1];
      hr_ = __builtin_amdgcn_wmma_f32_16x16x32_bf16(false, ah.v, false, b.v, (short)0, hr_, false, false); }
    { const uint4* p = (const uint4*)(Whh + (size_t)(col + 256) * DD + kk);
      b.q[0] = p[0]; b.q[1] = p[1];
      hz = __builtin_amdgcn_wmma_f32_16x16x32_bf16(false, ah.v, false, b.v, (short)0, hz, false, false); }
    { const uint4* p = (const uint4*)(Whh + (size_t)(col + 512) * DD + kk);
      b.q[0] = p[0]; b.q[1] = p[1];
      hn = __builtin_amdgcn_wmma_f32_16x16x32_bf16(false, ah.v, false, b.v, (short)0, hn, false, false); }
  }
  float bir = b_ih[col], biz = b_ih[col + 256], bin = b_ih[col + 512];
  float bhr = b_hh[col], bhz = b_hh[col + 256], bhn = b_hh[col + 512];
  int gbase = rowband * 16 + half * 8;
#pragma unroll
  for (int r = 0; r < 8; ++r) {
    int grow = gbase + r;
    float rg = sigmoidf_(ir[r] + bir + hr_[r] + bhr);
    float zg = sigmoidf_(iz[r] + biz + hz[r] + bhz);
    float ng = tanhf(in_[r] + bin + rg * (hn[r] + bhn));
    float hold = hin[(size_t)grow * DD + col];
    float hnew = (1.f - zg) * ng + zg * hold;
    hout[(size_t)grow * DD + col] = hnew;
    WT[(size_t)col * DD + grow] = f32_to_bf16(hnew);
  }
}

// ---------------------------------------------------------------------------
// y = x @ W_new ; one 16x16 tile per wave; emits yT_bf16[col][row].
// ---------------------------------------------------------------------------
__global__ void __launch_bounds__(256)
y_kernel(const float* __restrict__ x,
         const unsigned short* __restrict__ WT,
         unsigned short* __restrict__ yT) {
  __shared__ unsigned short lx[16 * 264];
  const int tid = threadIdx.x;
  const int rowband = blockIdx.x >> 1;   // 0..255 (16 rows each)
  const int colgrp  = blockIdx.x & 1;    // 0..1 (128 cols each)
  {
    int row = tid >> 4;
    int kb  = (tid & 15) * 16;
    const float* xr = x + (size_t)(rowband * 16 + row) * DD + kb;
    for (int i = 0; i < 16; i += 4) {
      float4 xv = *(const float4*)(xr + i);
      *(uint2*)&lx[row * 264 + kb + i] =
          make_uint2(pack_bf16x2(xv.x, xv.y), pack_bf16x2(xv.z, xv.w));
    }
  }
  __syncthreads();
  const int wave = tid >> 5, lane = tid & 31;
  const int mloc = lane & 15, half = lane >> 4;
  const int col = colgrp * 128 + wave * 16 + mloc;
  v8f acc = {};
  for (int kc = 0; kc < DD; kc += 32) {
    FragAB a, b;
    const unsigned short* pa = &lx[mloc * 264 + kc + half * 16];
    a.q[0] = *(const uint4*)pa; a.q[1] = *(const uint4*)(pa + 8);
    const uint4* p = (const uint4*)(WT + (size_t)col * DD + kc + half * 16);
    b.q[0] = p[0]; b.q[1] = p[1];
    acc = __builtin_amdgcn_wmma_f32_16x16x32_bf16(false, a.v, false, b.v, (short)0, acc, false, false);
  }
  // D-tile rows r+8*half are consecutive -> one packed b128 transposed store
  int mbase = rowband * 16 + half * 8;
  uint4 st;
  st.x = pack_bf16x2(acc[0], acc[1]); st.y = pack_bf16x2(acc[2], acc[3]);
  st.z = pack_bf16x2(acc[4], acc[5]); st.w = pack_bf16x2(acc[6], acc[7]);
  *(uint4*)(yT + (size_t)col * NN + mbase) = st;
}

// ---------------------------------------------------------------------------
// out = A @ y : HBM-bound GEMM. Streams fp32 A once (67 MB/step).
// KT=64 stage; async global->LDS double-buffered when available, else
// synchronous VGPR-staged bf16 LDS. B from L2-resident yT (2x b128/frag).
// ---------------------------------------------------------------------------
__global__ void __launch_bounds__(256)
out_kernel(const float* __restrict__ A,
           const unsigned short* __restrict__ yT,
           float* __restrict__ out) {
  const int tid = threadIdx.x;
  const int rowbase = blockIdx.x * 32;
  const int lrow = tid >> 3;          // 0..31
  const int kq = (tid & 7) * 8;       // 0..56 (8 floats / thread)
  const int wave = tid >> 5, lane = tid & 31;
  const int m0 = (wave >> 2) * 16, cg = wave & 3;
  const int mloc = lane & 15, half = lane >> 4;
  const int arow = m0 + mloc;
  v8f acc[4] = {};

#if USE_ASYNC
  __shared__ float laf[2][32 * 72];   // fp32 staged, double-buffered
  const float* gsrc = A + (size_t)(rowbase + lrow) * NN + kq;
  {
    float* ld = &laf[0][lrow * 72 + kq];
    __builtin_amdgcn_global_load_async_to_lds_b128((as1_v4i*)gsrc, (as3_v4i*)ld, 0, 0);
    __builtin_amdgcn_global_load_async_to_lds_b128((as1_v4i*)(gsrc + 4), (as3_v4i*)(ld + 4), 0, 0);
  }
  for (int k0 = 0; k0 < NN; k0 += 64) {
    const int cur = (k0 >> 6) & 1;
    if (k0 + 64 < NN) {
      float* ld = &laf[cur ^ 1][lrow * 72 + kq];
      __builtin_amdgcn_global_load_async_to_lds_b128((as1_v4i*)(gsrc + k0 + 64), (as3_v4i*)ld, 0, 0);
      __builtin_amdgcn_global_load_async_to_lds_b128((as1_v4i*)(gsrc + k0 + 68), (as3_v4i*)(ld + 4), 0, 0);
      __builtin_amdgcn_s_wait_asynccnt(2);   // previous chunk's 2 loads done
    } else {
      __builtin_amdgcn_s_wait_asynccnt(0);
    }
    __syncthreads();
#pragma unroll
    for (int ks = 0; ks < 64; ks += 32) {
      FragAB a;
      const float* pa = &laf[cur][arow * 72 + ks + half * 16];
      cvt8(pa, a.q[0]);
      cvt8(pa + 8, a.q[1]);
#pragma unroll
      for (int j = 0; j < 4; ++j) {
        int colj = (cg * 4 + j) * 16 + mloc;
        FragAB b;
        const uint4* p = (const uint4*)(yT + (size_t)colj * NN + k0 + ks + half * 16);
        b.q[0] = p[0]; b.q[1] = p[1];
        acc[j] = __builtin_amdgcn_wmma_f32_16x16x32_bf16(false, a.v, false, b.v, (short)0, acc[j], false, false);
      }
    }
    __syncthreads();
  }
#else
  __shared__ unsigned short la[32 * 72];   // bf16 staged
  const float* ar = A + (size_t)(rowbase + lrow) * NN + kq;
  for (int k0 = 0; k0 < NN; k0 += 64) {
    float4 a0 = *(const float4*)(ar + k0);
    float4 a1 = *(const float4*)(ar + k0 + 4);
    uint4 st;
    st.x = pack_bf16x2(a0.x, a0.y); st.y = pack_bf16x2(a0.z, a0.w);
    st.z = pack_bf16x2(a1.x, a1.y); st.w = pack_bf16x2(a1.z, a1.w);
    *(uint4*)&la[lrow * 72 + kq] = st;
    __syncthreads();
#pragma unroll
    for (int ks = 0; ks < 64; ks += 32) {
      FragAB a;
      const unsigned short* pa = &la[arow * 72 + ks + half * 16];
      a.q[0] = *(const uint4*)pa; a.q[1] = *(const uint4*)(pa + 8);
#pragma unroll
      for (int j = 0; j < 4; ++j) {
        int colj = (cg * 4 + j) * 16 + mloc;
        FragAB b;
        const uint4* p = (const uint4*)(yT + (size_t)colj * NN + k0 + ks + half * 16);
        b.q[0] = p[0]; b.q[1] = p[1];
        acc[j] = __builtin_amdgcn_wmma_f32_16x16x32_bf16(false, a.v, false, b.v, (short)0, acc[j], false, false);
      }
    }
    __syncthreads();
  }
#endif

#pragma unroll
  for (int j = 0; j < 4; ++j) {
    int colj = (cg * 4 + j) * 16 + mloc;
    int gm = rowbase + m0 + half * 8;
#pragma unroll
    for (int r = 0; r < 8; ++r)
      out[(size_t)(gm + r) * DD + colj] = acc[j][r];
  }
}

// ---------------------------------------------------------------------------
extern "C" void kernel_launch(void* const* d_in, const int* in_sizes, int n_in,
                              void* d_out, int out_size, void* d_ws, size_t ws_size,
                              hipStream_t stream) {
  const float* x_seq  = (const float*)d_in[0];
  const float* A_seq  = (const float*)d_in[1];
  const float* pool_p = (const float*)d_in[2];
  const float* W_ih   = (const float*)d_in[3];
  const float* W_hh   = (const float*)d_in[4];
  const float* b_ih   = (const float*)d_in[5];
  const float* b_hh   = (const float*)d_in[6];
  const float* W0     = (const float*)d_in[7];
  float* out = (float*)d_out;

  char* ws = (char*)d_ws;
  float*          s      = (float*)(ws + 0);                  // 16 KB
  int*            sel    = (int*)(ws + 16384);                // 1 KB
  float*          gate   = (float*)(ws + 17408);              // 1 KB
  float*          pninv  = (float*)(ws + 18432);              // 256 B
  float*          h0     = (float*)(ws + 18688);              // 256 KB
  float*          h1     = (float*)(ws + 280832);             // 256 KB
  unsigned short* WT     = (unsigned short*)(ws + 542976);    // 128 KB
  unsigned short* Wih_bf = (unsigned short*)(ws + 674048);    // 384 KB
  unsigned short* Whh_bf = (unsigned short*)(ws + 1067264);   // 384 KB
  unsigned short* yT     = (unsigned short*)(ws + 1460480);   // 2 MB
  (void)in_sizes; (void)n_in; (void)out_size; (void)ws_size;

  prep_kernel<<<768, 256, 0, stream>>>(pool_p, W_ih, W_hh, W0, pninv, Wih_bf, Whh_bf, h0);

  for (int t = 0; t < T_STEPS; ++t) {
    const float* x = x_seq + (size_t)t * NN * DD;
    const float* A = A_seq + (size_t)t * NN * NN;
    const float* hin  = (t & 1) ? h1 : h0;
    float*       hout = (t & 1) ? h0 : h1;
    score_kernel<<<512, 256, 0, stream>>>(x, pool_p, pninv, s);
    rank_kernel<<<16, 256, 0, stream>>>(s, sel, gate);
    gru_kernel<<<32, 256, 0, stream>>>(x, sel, gate, Wih_bf, Whh_bf, b_ih, b_hh, hin, hout, WT);
    y_kernel<<<512, 256, 0, stream>>>(x, WT, yT);
    out_kernel<<<128, 256, 0, stream>>>(A, yT, out + (size_t)t * NN * DD);
  }
}
